// Attention_7086696038697
// MI455X (gfx1250) — matrix operand order repaired and checked by reading, not compiled, not run on hardware
//
#include <hip/hip_runtime.h>
#include <cmath>

// MI455X attention: bf16 WMMA (16x16x32, f32 acc) for QKV projections and
// flash-style attention, both software-pipelined (double-buffered LDS).
// Workspace: Qb/Kb/Vb bf16 [8192 x 1024] each = 16MB each, 48MB total.

typedef __attribute__((ext_vector_type(16))) __bf16 v16bf;
typedef __attribute__((ext_vector_type(8)))  __bf16 v8bf;
typedef __attribute__((ext_vector_type(4)))  __bf16 v4bf;
typedef __attribute__((ext_vector_type(8)))  float  v8f;
typedef __attribute__((ext_vector_type(4)))  float  v4f;

static __device__ __forceinline__ v16bf join8(v8bf a, v8bf b) {
  v16bf r;
#pragma unroll
  for (int i = 0; i < 8; ++i) { r[i] = a[i]; r[i + 8] = b[i]; }
  return r;
}

static __device__ __forceinline__ v8f wmma_bf16(v16bf a, v16bf b, v8f c) {
  return __builtin_amdgcn_wmma_f32_16x16x32_bf16(false, a, false, b,
                                                 (short)0, c, false, false);
}

// global -> LDS async copy, 16B per lane (gfx1250 GLOBAL_LOAD_ASYNC_TO_LDS_B128)
static __device__ __forceinline__ void async_copy_b128(const void* gsrc, void* ldst) {
  unsigned lds = (unsigned)(unsigned long long)ldst;  // ADDR[31:0] = LDS offset
  asm volatile("global_load_async_to_lds_b128 %0, %1, off"
               :: "v"(lds), "v"((unsigned long long)gsrc)
               : "memory");
}
static __device__ __forceinline__ void async_wait0() {
  asm volatile("s_wait_asynccnt 0x0" ::: "memory");
}

// ---------------------------------------------------------------------------
// Phase 1: Y[8192,1024](bf16) = scale * X[8192,1024](f32) @ W[1024,1024](f32)
// Block = 256 threads (8 waves), 128-row x 64-col tile, double-buffered LDS.
// ---------------------------------------------------------------------------
__global__ __launch_bounds__(256) void proj_kernel(
    const float* __restrict__ X, const float* __restrict__ Wt,
    unsigned short* __restrict__ Yo, float scale) {
  __bf16* __restrict__ Y = (__bf16*)Yo;
  constexpr int ASTR = 40, ABUF = 128 * ASTR;  // 80B rows, 16B-aligned runs
  constexpr int BSTR = 40, BBUF = 64 * BSTR;
  __shared__ __align__(32) __bf16 As[2 * ABUF];   // [buf][row][k]  20KB
  __shared__ __align__(32) __bf16 Bst[2 * BBUF];  // [buf][col][k]  10KB

  const int t = threadIdx.x;
  const int lane = t & 31, wv = t >> 5;
  const int hi = lane >> 4, ln = lane & 15;
  const int r0 = blockIdx.x * 128;
  const int c0 = blockIdx.y * 64;

  // staging decomposition (16B per lane)
  const int ar_ = t >> 3, ak4 = (t & 7) * 4;          // A: 32 rows x 8 float4 / 256t x4
  const int bk_ = t >> 4, bc4 = (t & 15) * 4;         // B: 32 rows x 16 float4 / 256t x2

  auto loadA = [&](int kc, v4f* xa) {
#pragma unroll
    for (int e = 0; e < 4; ++e)
      xa[e] = *(const v4f*)&X[(r0 + ar_ + 32 * e) * 1024 + kc + ak4];
  };
  auto loadB = [&](int kc, v4f* wb) {
#pragma unroll
    for (int e = 0; e < 2; ++e)
      wb[e] = *(const v4f*)&Wt[(kc + bk_ + 16 * e) * 1024 + c0 + bc4];
  };
  auto stageLDS = [&](int buf, const v4f* xa, const v4f* wb) {
#pragma unroll
    for (int e = 0; e < 4; ++e) {
      v4bf y = {(__bf16)xa[e][0], (__bf16)xa[e][1], (__bf16)xa[e][2], (__bf16)xa[e][3]};
      *(v4bf*)&As[buf * ABUF + (ar_ + 32 * e) * ASTR + ak4] = y;
    }
#pragma unroll
    for (int e = 0; e < 2; ++e)
#pragma unroll
      for (int i = 0; i < 4; ++i)
        Bst[buf * BBUF + (bc4 + i) * BSTR + bk_ + 16 * e] = (__bf16)wb[e][i];
  };

  v8f acc[4] = {};
  {  // prologue: stage k-tile 0 into buffer 0
    v4f xa[4], wb[2];
    loadA(0, xa);
    loadB(0, wb);
    stageLDS(0, xa, wb);
  }
  __syncthreads();

  for (int i = 0; i < 32; ++i) {
    const int cur = i & 1;
    v4f xa[4], wb[2];
    if (i < 31) {  // issue next tile's global loads; latency hidden by WMMAs
      loadA((i + 1) * 32, xa);
      loadB((i + 1) * 32, wb);
      if (i < 30) __builtin_prefetch(&X[(r0 + (t >> 1)) * 1024 + (i + 2) * 32], 0, 0);
    }

    // A-frag (16x32): lane = row M, two 8-elem runs at k = 8hi and 8hi+16
    const __bf16* arow = &As[cur * ABUF + (16 * wv + ln) * ASTR + 8 * hi];
    v16bf af = join8(*(const v8bf*)arow, *(const v8bf*)(arow + 16));
#pragma unroll
    for (int tt = 0; tt < 4; ++tt) {
      // B-frag (32x16): lane = col N, two 8-elem runs at k = 8hi and 8hi+16
      const __bf16* br = &Bst[cur * BBUF + (16 * tt + ln) * BSTR + 8 * hi];
      v16bf bfr = join8(*(const v8bf*)br, *(const v8bf*)(br + 16));
      acc[tt] = wmma_bf16(af, bfr, acc[tt]);
    }

    if (i < 31) stageLDS(1 - cur, xa, wb);  // fill spare buffer
    __syncthreads();                        // one barrier/iter: reuse distance 2
  }

  // C-frag layout: lane holds N=ln, VGPR v holds M = v + 8*hi
#pragma unroll
  for (int tt = 0; tt < 4; ++tt)
#pragma unroll
    for (int v = 0; v < 8; ++v) {
      int row = r0 + 16 * wv + v + 8 * hi;
      int col = c0 + 16 * tt + ln;
      Y[row * 1024 + col] = (__bf16)(acc[tt][v] * scale);
    }
}

// ---------------------------------------------------------------------------
// Phase 2: flash attention per (b, h). 8 waves/block, each owns a 16-row
// q-tile. S^T = K_tile @ Q^T puts q along lanes (per-lane softmax) and the
// exp'd P lands directly in the P@V A-operand layout. K tiles stream through
// GLOBAL_LOAD_ASYNC_TO_LDS_B128 into the spare LDS buffer while WMMAs run.
// ---------------------------------------------------------------------------
__global__ __launch_bounds__(256) void attn_kernel(
    const unsigned short* __restrict__ Qp, const unsigned short* __restrict__ Kp,
    const unsigned short* __restrict__ Vp, float* __restrict__ Out) {
  const __bf16* __restrict__ Qb = (const __bf16*)Qp;
  const __bf16* __restrict__ Kb = (const __bf16*)Kp;
  const __bf16* __restrict__ Vb = (const __bf16*)Vp;
  constexpr int KSTR = 80, KBUF = 32 * KSTR;  // 160B rows
  constexpr int VSTR = 40, VBUF = 64 * VSTR;  //  80B rows
  __shared__ __align__(32) __bf16 Ks[2 * KBUF];  // K tile [m][k]            10KB
  __shared__ __align__(32) __bf16 Vt[2 * VBUF];  // V tile transposed [v][m] 10KB

  const int t = threadIdx.x;
  const int lane = t & 31, wv = t >> 5;
  const int hi = lane >> 4, ln = lane & 15;
  const int h = blockIdx.y, b = blockIdx.z;
  const int q0 = blockIdx.x * 128 + wv * 16;

  const int krow = t >> 3, koff = (t & 7) * 8;  // staging: 16B per lane

  // Q B-fragments (kept in VGPRs): lane = q column, two 8-elem runs at k = 8hi and 8hi+16
  const __bf16* qrow = Qb + (size_t)(b * 1024 + q0 + ln) * 1024 + h * 64;
  v16bf qf[2];
#pragma unroll
  for (int kc2 = 0; kc2 < 2; ++kc2) {
    const __bf16* p = qrow + kc2 * 32 + 8 * hi;
    qf[kc2] = join8(*(const v8bf*)p, *(const v8bf*)(p + 16));
  }

  auto stageK = [&](int m0, int buf) {
    async_copy_b128(Kb + (size_t)(b * 1024 + m0 + krow) * 1024 + h * 64 + koff,
                    &Ks[buf * KBUF + krow * KSTR + koff]);
  };
  auto loadV = [&](int m0) {
    return *(const v8bf*)(Vb + (size_t)(b * 1024 + m0 + krow) * 1024 + h * 64 + koff);
  };
  auto storeVt = [&](int buf, v8bf vv) {
#pragma unroll
    for (int i = 0; i < 8; ++i) Vt[buf * VBUF + (koff + i) * VSTR + krow] = vv[i];
  };

  v8f o[4] = {};
  float mx = -1e30f, l = 0.f;

  {  // prologue: stage m-tile 0 into buffer 0
    stageK(0, 0);
    storeVt(0, loadV(0));
    async_wait0();
  }
  __syncthreads();

  for (int i = 0; i < 32; ++i) {
    const int cur = i & 1;
    const int m0n = (i + 1) * 32;
    v8bf vn;
    if (i < 31) {          // async copy of next K tile runs during the WMMAs
      stageK(m0n, 1 - cur);
      vn = loadV(m0n);
    }

    // S^T[m][q]: C-frag -> lane holds q=ln, VGPR v holds m = 8*hi + v (+16 for sc[1])
    v8f sc[2] = {{0, 0, 0, 0, 0, 0, 0, 0}, {0, 0, 0, 0, 0, 0, 0, 0}};
#pragma unroll
    for (int ms = 0; ms < 2; ++ms)
#pragma unroll
      for (int kc2 = 0; kc2 < 2; ++kc2) {
        const __bf16* ar = &Ks[cur * KBUF + (16 * ms + ln) * KSTR + kc2 * 32 + 8 * hi];
        v16bf af = join8(*(const v8bf*)ar, *(const v8bf*)(ar + 16));
        sc[ms] = wmma_bf16(af, qf[kc2], sc[ms]);
      }

    // online softmax (exp2 domain; 0.125*log2e pre-folded into Q)
    float tmax = sc[0][0];
#pragma unroll
    for (int v = 1; v < 8; ++v) tmax = fmaxf(tmax, sc[0][v]);
#pragma unroll
    for (int v = 0; v < 8; ++v) tmax = fmaxf(tmax, sc[1][v]);
    tmax = fmaxf(tmax, __shfl_xor(tmax, 16, 32));  // other half-wave's m values
    float nm = fmaxf(mx, tmax);
    float alpha = exp2f(mx - nm);
    mx = nm;
    float p0[8], p1[8], ls = 0.f;
#pragma unroll
    for (int v = 0; v < 8; ++v) {
      p0[v] = exp2f(sc[0][v] - mx);
      p1[v] = exp2f(sc[1][v] - mx);
      ls += p0[v] + p1[v];
    }
    ls += __shfl_xor(ls, 16, 32);
    l = l * alpha + ls;

    // rescale O rows: q for VGPR v is v + 8*hi; alpha lives at lane q
#pragma unroll
    for (int v = 0; v < 8; ++v) {
      float av = __shfl(alpha, v + 8 * hi, 32);
      o[0][v] *= av; o[1][v] *= av; o[2][v] *= av; o[3][v] *= av;
    }

    // P lands exactly in the A-operand layout: e[i<8]=sc0 regs, e[i>=8]=sc1 regs
    v16bf pa;
#pragma unroll
    for (int i2 = 0; i2 < 8; ++i2) { pa[i2] = (__bf16)p0[i2]; pa[i2 + 8] = (__bf16)p1[i2]; }

#pragma unroll
    for (int tt = 0; tt < 4; ++tt) {
      const __bf16* br = &Vt[cur * VBUF + (16 * tt + ln) * VSTR + 8 * hi];
      v16bf vf = join8(*(const v8bf*)br, *(const v8bf*)(br + 16));
      o[tt] = wmma_bf16(pa, vf, o[tt]);
    }

    if (i < 31) storeVt(1 - cur, vn);  // fill spare V buffer
    async_wait0();                     // next K tile landed in LDS
    __syncthreads();                   // one barrier/iter: reuse distance 2
  }

  // finalize: out[b, q, h, vcol], q = q0 + v + 8*hi, vcol = 16*tt + ln
#pragma unroll
  for (int v = 0; v < 8; ++v) {
    float lv = __shfl(l, v + 8 * hi, 32);
    float inv = 1.0f / lv;
    int q = q0 + v + 8 * hi;
    float* orow = Out + (size_t)((b * 1024 + q) * 16 + h) * 64;
#pragma unroll
    for (int tt = 0; tt < 4; ++tt) orow[16 * tt + ln] = o[tt][v] * inv;
  }
}

// ---------------------------------------------------------------------------
extern "C" void kernel_launch(void* const* d_in, const int* in_sizes, int n_in,
                              void* d_out, int out_size, void* d_ws, size_t ws_size,
                              hipStream_t stream) {
  (void)in_sizes; (void)n_in; (void)out_size; (void)ws_size;
  const float* X  = (const float*)d_in[0];
  const float* Wq = (const float*)d_in[1];
  const float* Wk = (const float*)d_in[2];
  const float* Wv = (const float*)d_in[3];
  float* Out = (float*)d_out;

  unsigned short* Qb = (unsigned short*)d_ws;                 // 16MB
  unsigned short* Kb = Qb + (size_t)8192 * 1024;              // 16MB
  unsigned short* Vb = Kb + (size_t)8192 * 1024;              // 16MB

  const float scaleQ = 0.125f * 1.44269504088896340736f;  // kdim^-0.5 * log2(e)
  dim3 g1(64, 16), b1(256);
  proj_kernel<<<g1, b1, 0, stream>>>(X, Wq, Qb, scaleQ);
  proj_kernel<<<g1, b1, 0, stream>>>(X, Wk, Kb, 1.0f);
  proj_kernel<<<g1, b1, 0, stream>>>(X, Wv, Vb, 1.0f);

  dim3 g2(8, 16, 8), b2(256);
  attn_kernel<<<g2, b2, 0, stream>>>(Qb, Kb, Vb, Out);
}
